// HeterSUMGraph_12395275616366
// MI455X (gfx1250) — matrix operand order repaired and checked by reading, not compiled, not run on hardware
//
#include <hip/hip_runtime.h>
#include <hip/hip_bf16.h>
#include <math.h>

// ---------------------------------------------------------------------------
// Problem constants (from reference)
// ---------------------------------------------------------------------------
#define NWRD 15000   // words
#define NSNT 2000    // sentences
#define SLEN 50      // tokens per sentence
#define NEDG 40000   // edges
#define NHEAD 8
#define EMBD 300     // word embedding dim
#define EPSF 1e-5f

typedef __bf16 bf16;
typedef __attribute__((ext_vector_type(16))) __bf16 v16bf;
typedef __attribute__((ext_vector_type(8)))  __bf16 v8bf;
typedef __attribute__((ext_vector_type(8)))  float  v8f;

// ---------------------------------------------------------------------------
// Small device helpers
// ---------------------------------------------------------------------------
__device__ __forceinline__ float sigm(float x) { return 1.0f / (1.0f + expf(-x)); }

__device__ __forceinline__ float wave_reduce(float v) {
  #pragma unroll
  for (int o = 16; o > 0; o >>= 1) v += __shfl_down(v, o, 32);
  return v;
}

__device__ float block_reduce_sum(float v) {
  __shared__ float sh[256];
  sh[threadIdx.x] = v;
  __syncthreads();
  for (int s = 128; s > 0; s >>= 1) {
    if ((int)threadIdx.x < s) sh[threadIdx.x] += sh[threadIdx.x + s];
    __syncthreads();
  }
  float r = sh[0];
  __syncthreads();
  return r;
}

// monotone float<->uint key for atomic max on floats (incl. negatives)
__device__ __forceinline__ unsigned fkey(float f) {
  unsigned u = __float_as_uint(f);
  return (u & 0x80000000u) ? ~u : (u | 0x80000000u);
}
__device__ __forceinline__ float fdec(unsigned k) {
  return (k & 0x80000000u) ? __uint_as_float(k & 0x7fffffffu) : __uint_as_float(~k);
}

__device__ __forceinline__ v16bf cat16(v8bf lo, v8bf hi) {
  return __builtin_shufflevector(lo, hi, 0, 1, 2, 3, 4, 5, 6, 7,
                                 8, 9, 10, 11, 12, 13, 14, 15);
}

// ---------------------------------------------------------------------------
// WMMA bf16 GEMM:  C[M,N] = A[M,K] @ W[N,K]^T (+bias[n]) (+optional ReLU)
//
// - A is (M x K) with leading dim lda; W is (N x K) with leading dim ldw.
// - lda/ldw are multiples of 32 and the pad K..ld is ZERO-FILLED by the
//   staging kernels, so the K loop has no guards and every sub-load is an
//   aligned 16B global_load_b128.
// - Row indices are clamped (min(row, M-1)): phantom rows/cols are computed
//   but never stored, so all stored elements are exact.
// - One wave computes a 16x64 tile (1 A fragment reused over 4 B subtiles,
//   4 v_wmma_f32_16x16x32_bf16 per k-step). 8 waves per block.
// ---------------------------------------------------------------------------
template <int RELU>
__global__ __launch_bounds__(256) void gemm_bt_wmma(
    const bf16* __restrict__ A, int lda,
    const bf16* __restrict__ W, int ldw,
    const float* __restrict__ bias, float* __restrict__ C,
    int M, int N, int K) {
  int wave = threadIdx.x >> 5;
  int tm = (M + 15) >> 4;
  long tn4 = (N + 63) >> 6;
  long tile = (long)blockIdx.x * 8 + wave;
  if (tile >= (long)tm * tn4) return;  // whole-wave uniform exit (EXEC all-1 for WMMA)
  int tmi = (int)(tile % tm);
  int tni = (int)(tile / tm);
  int m0 = tmi << 4, n0 = tni << 6;

  int lane = threadIdx.x & 31;
  int half = lane >> 4;
  int li   = lane & 15;

  const bf16* pa  = A + (size_t)min(m0 + li, M - 1) * lda + half * 16;
  const bf16* pb0 = W + (size_t)min(n0 + li,      N - 1) * ldw + half * 16;
  const bf16* pb1 = W + (size_t)min(n0 + 16 + li, N - 1) * ldw + half * 16;
  const bf16* pb2 = W + (size_t)min(n0 + 32 + li, N - 1) * ldw + half * 16;
  const bf16* pb3 = W + (size_t)min(n0 + 48 + li, N - 1) * ldw + half * 16;

  v8f acc0 = {0.f, 0.f, 0.f, 0.f, 0.f, 0.f, 0.f, 0.f};
  v8f acc1 = acc0, acc2 = acc0, acc3 = acc0;

  int kPad = (K + 31) & ~31;
  for (int k0 = 0; k0 < kPad; k0 += 32) {
    // A operand, this code's contiguous k order: lane regs hold K = {h*16..h*16+15}
    v16bf a = cat16(*(const v8bf*)(pa + k0), *(const v8bf*)(pa + k0 + 8));
    // B operand, same contiguous k order as A: lane = N column, half selects contiguous K block of 16
    v16bf b0 = cat16(*(const v8bf*)(pb0 + k0), *(const v8bf*)(pb0 + k0 + 8));
    acc0 = __builtin_amdgcn_wmma_f32_16x16x32_bf16(false, a, false, b0, (short)0, acc0, false, false);
    v16bf b1 = cat16(*(const v8bf*)(pb1 + k0), *(const v8bf*)(pb1 + k0 + 8));
    acc1 = __builtin_amdgcn_wmma_f32_16x16x32_bf16(false, a, false, b1, (short)0, acc1, false, false);
    v16bf b2 = cat16(*(const v8bf*)(pb2 + k0), *(const v8bf*)(pb2 + k0 + 8));
    acc2 = __builtin_amdgcn_wmma_f32_16x16x32_bf16(false, a, false, b2, (short)0, acc2, false, false);
    v16bf b3 = cat16(*(const v8bf*)(pb3 + k0), *(const v8bf*)(pb3 + k0 + 8));
    acc3 = __builtin_amdgcn_wmma_f32_16x16x32_bf16(false, a, false, b3, (short)0, acc3, false, false);
  }

  // C/D layout: reg r, lanes 0-15 -> M=r,N=lane ; lanes 16-31 -> M=8+r
  #define STORE_SUB(ACC, S)                                              \
    {                                                                    \
      int nn = n0 + (S) * 16 + li;                                       \
      if (nn < N) {                                                      \
        float bb = bias ? bias[nn] : 0.0f;                               \
        _Pragma("unroll")                                                \
        for (int r = 0; r < 8; ++r) {                                    \
          int mm = m0 + half * 8 + r;                                    \
          if (mm < M) {                                                  \
            float v = ACC[r] + bb;                                       \
            if (RELU) v = fmaxf(v, 0.0f);                                \
            C[(size_t)mm * N + nn] = v;                                  \
          }                                                              \
        }                                                                \
      }                                                                  \
    }
  STORE_SUB(acc0, 0)
  STORE_SUB(acc1, 1)
  STORE_SUB(acc2, 2)
  STORE_SUB(acc3, 3)
  #undef STORE_SUB
}

// ---------------------------------------------------------------------------
// Staging kernels: cast/gather f32 -> bf16 with zero-filled K padding
// ---------------------------------------------------------------------------
__global__ void cast_pad_k(const float* __restrict__ s, bf16* __restrict__ d,
                           int rows, int K, int ld) {
  long i = (long)blockIdx.x * 256 + threadIdx.x;
  if (i >= (long)rows * ld) return;
  int r = (int)(i / ld), c = (int)(i % ld);
  d[i] = (c < K) ? (bf16)s[(size_t)r * K + c] : (bf16)0.0f;
}
__global__ void gather_pad_bf16_k(const int* __restrict__ idx, const float* __restrict__ tab,
                                  bf16* __restrict__ out, int rows, int D, int ld) {
  long i = (long)blockIdx.x * 256 + threadIdx.x;
  if (i >= (long)rows * ld) return;
  int r = (int)(i / ld), c = (int)(i % ld);
  out[i] = (c < D) ? (bf16)tab[(size_t)idx[r] * D + c] : (bf16)0.0f;
}
__global__ void gather_rows_f32_k(const int* __restrict__ idx, const float* __restrict__ tab,
                                  float* __restrict__ out, int rows, int D) {
  long i = (long)blockIdx.x * 256 + threadIdx.x;
  if (i >= (long)rows * D) return;
  int r = (int)(i / D), c = (int)(i % D);
  out[i] = tab[(size_t)idx[r] * D + c];
}
__global__ void gather_edge_k(const float* __restrict__ eraw, const float* __restrict__ etab,
                              float* __restrict__ out) {
  long i = (long)blockIdx.x * 256 + threadIdx.x;
  if (i >= (long)NEDG * 50) return;
  int e = (int)(i / 50), k = (int)(i % 50);
  int id = (int)floorf(eraw[e] * 100.0f);
  id = max(0, min(id, 100));
  out[i] = etab[id * 50 + k];
}

__global__ void fill_f32_k(float* p, float v, long n) {
  long i = (long)blockIdx.x * 256 + threadIdx.x;
  if (i < n) p[i] = v;
}
__global__ void fill_u32_k(unsigned* p, unsigned v, long n) {
  long i = (long)blockIdx.x * 256 + threadIdx.x;
  if (i < n) p[i] = v;
}
__global__ void fill_key_k(unsigned* p, long n) {
  long i = (long)blockIdx.x * 256 + threadIdx.x;
  if (i < n) p[i] = fkey(-3.0e38f);
}

// LayerNorm (no affine), one row per block (blockDim=256)
__global__ void layernorm_k(float* __restrict__ x, int D) {
  float* p = x + (size_t)blockIdx.x * D;
  float s = 0.f;
  for (int i = threadIdx.x; i < D; i += 256) s += p[i];
  float mu = block_reduce_sum(s) / (float)D;
  float v = 0.f;
  for (int i = threadIdx.x; i < D; i += 256) { float d = p[i] - mu; v += d * d; }
  float var = block_reduce_sum(v) / (float)D;
  float inv = rsqrtf(var + EPSF);
  for (int i = threadIdx.x; i < D; i += 256) p[i] = (p[i] - mu) * inv;
}

// ---------------------------------------------------------------------------
// CNN path
// ---------------------------------------------------------------------------
__global__ void conv1_embed_k(const int* __restrict__ Xs, const float* __restrict__ emb,
                              const float* __restrict__ w, const float* __restrict__ b,
                              float* __restrict__ y) {
  int np = blockIdx.x;  // n*SLEN + pos
  int n = np / SLEN, pos = np % SLEN;
  int oc = threadIdx.x;
  if (oc >= 100) return;
  float acc = b[oc];
  for (int kw = 0; kw < 3; ++kw) {
    int pi = pos + kw - 1;
    if (pi < 0 || pi >= SLEN) continue;
    int tok = Xs[n * SLEN + pi];
    const float* e  = emb + (size_t)tok * EMBD;
    const float* ww = w + ((size_t)oc * EMBD) * 3 + kw;  // (O,I,H) layout
    float a = 0.f;
    for (int ic = 0; ic < EMBD; ++ic) a += e[ic] * ww[(size_t)ic * 3];
    acc += a;
  }
  y[((size_t)n * 100 + oc) * SLEN + pos] = acc;
}

__global__ void conv1d_g_k(const float* __restrict__ x, const float* __restrict__ w,
                           const float* __restrict__ b, float* __restrict__ y,
                           int N, int Cin, int Cout, int L, int Kw, int padLo) {
  int np = blockIdx.x;
  int n = np / L, pos = np % L;
  int oc = threadIdx.x;
  if (oc >= Cout) return;
  float acc = b[oc];
  for (int kw = 0; kw < Kw; ++kw) {
    int pi = pos + kw - padLo;
    if (pi < 0 || pi >= L) continue;
    const float* xp = x + ((size_t)n * Cin) * L + pi;
    const float* wp = w + ((size_t)oc * Cin) * Kw + kw;
    for (int ic = 0; ic < Cin; ++ic) acc += xp[(size_t)ic * L] * wp[(size_t)ic * Kw];
  }
  y[((size_t)n * Cout + oc) * L + pos] = acc;
}

__global__ void bn_stats_k(const float* __restrict__ x, int N, int C, int L,
                           float* __restrict__ m, float* __restrict__ v) {
  int ch = blockIdx.x;
  float s = 0.f, q = 0.f;
  for (int t = threadIdx.x; t < N * L; t += 256) {
    int n = t / L, l = t % L;
    float val = x[((size_t)n * C + ch) * L + l];
    s += val; q += val * val;
  }
  float S = block_reduce_sum(s);
  float Q = block_reduce_sum(q);
  if (threadIdx.x == 0) {
    float mu = S / (float)(N * L);
    m[ch] = mu;
    v[ch] = Q / (float)(N * L) - mu * mu;
  }
}

__global__ void bn_relu_k(float* __restrict__ x, const float* __restrict__ m,
                          const float* __restrict__ v, const float* __restrict__ g,
                          const float* __restrict__ b, int C, int L, long n) {
  long i = (long)blockIdx.x * 256 + threadIdx.x;
  if (i >= n) return;
  int ch = (int)((i / L) % C);
  float y = (x[i] - m[ch]) * rsqrtf(v[ch] + EPSF) * g[ch] + b[ch];
  x[i] = fmaxf(y, 0.0f);
}

__global__ void xscnn_max_k(const float* __restrict__ y3, float* __restrict__ Hs) {
  long i = (long)blockIdx.x * 256 + threadIdx.x;
  if (i >= (long)NSNT * 64) return;
  int n = (int)(i / 64), c = (int)(i % 64);
  const float* p = y3 + ((size_t)n * 64 + c) * SLEN;
  float m = p[0];
  for (int l = 1; l < SLEN; ++l) m = fmaxf(m, p[l]);
  Hs[n * 128 + c] = m;
}

// ---------------------------------------------------------------------------
// LSTM recurrence (hidden = 32). Zin already holds x@Wih^T + b.
// One block (128 thr) per sentence; accumulates masked mean into Hs[:,64+dir*32..]
// ---------------------------------------------------------------------------
__global__ __launch_bounds__(128) void lstm_k(const float* __restrict__ Zin,
                                              const float* __restrict__ Whh,
                                              const int* __restrict__ Xs,
                                              float* __restrict__ Hs, int reverse) {
  __shared__ float h[32], c[32], z[128];
  __shared__ int slen;
  int n = blockIdx.x;
  int j = threadIdx.x;
  if (j == 0) {
    int cnt = 0;
    for (int l = 0; l < SLEN; ++l) cnt += (Xs[n * SLEN + l] != 0);
    slen = max(cnt, 1);
  }
  if (j < 32) { h[j] = 0.f; c[j] = 0.f; }
  __syncthreads();
  float acc = 0.f;
  for (int t = 0; t < SLEN; ++t) {
    int tp = reverse ? (SLEN - 1 - t) : t;  // original position consumed this step
    float zj = Zin[((size_t)n * SLEN + tp) * 128 + j];
    #pragma unroll 4
    for (int u = 0; u < 32; ++u) zj += h[u] * Whh[j * 32 + u];
    z[j] = zj;
    __syncthreads();
    if (j < 32) {
      float ig = sigm(z[j]);
      float fg = sigm(z[32 + j]);
      float gg = tanhf(z[64 + j]);
      float og = sigm(z[96 + j]);
      float cn = fg * c[j] + ig * gg;
      float hn = og * tanhf(cn);
      c[j] = cn; h[j] = hn;
      if (tp < slen) acc += hn;  // masked-mean contribution (position tp)
    }
    __syncthreads();
  }
  if (j < 32) Hs[n * 128 + 64 + reverse * 32 + j] = acc / (float)slen;
}

// ---------------------------------------------------------------------------
// GATv2 kernels.  Extended edge list: e < E -> (src[e], dst[e]); else self-loop
// (e-E, e-E) with zero edge features.  eh = eattr @ We^T computed on the fly
// (never materialized: saves ~400 MB of HBM traffic).
// ---------------------------------------------------------------------------
__global__ void gat_logits_k(const float* __restrict__ xl, const float* __restrict__ xr,
                             const float* __restrict__ ea, const float* __restrict__ We,
                             const float* __restrict__ att,
                             const int* __restrict__ sidx, const int* __restrict__ didx,
                             int E, int C, float* __restrict__ logits,
                             unsigned* __restrict__ mxu) {
  long eh = blockIdx.x;
  int h = (int)(eh % NHEAD);
  long e = eh / NHEAD;
  int s, d;
  if (e < E) { s = sidx[e]; d = didx[e]; } else { s = d = (int)(e - E); }
  int lane = threadIdx.x;
  const float* pl = xl + (size_t)s * NHEAD * C + (size_t)h * C;
  const float* pr = xr + (size_t)d * NHEAD * C + (size_t)h * C;
  float part = 0.f;
  for (int cc = lane; cc < C; cc += 32) {
    float m = pl[cc] + pr[cc];
    if (e < E) {
      const float* we = We + ((size_t)(h * C + cc)) * 50;
      const float* ep = ea + (size_t)e * 50;
      float a = 0.f;
      #pragma unroll 10
      for (int k = 0; k < 50; ++k) a += ep[k] * we[k];
      m += a;
    }
    float g = (m > 0.f) ? m : 0.2f * m;  // leaky_relu(0.2)
    part += g * att[h * C + cc];
  }
  part = wave_reduce(part);
  if (lane == 0) {
    logits[eh] = part;
    atomicMax(&mxu[d * NHEAD + h], fkey(part));
  }
}

__global__ void gat_exp_k(const float* __restrict__ logits, const unsigned* __restrict__ mxu,
                          const int* __restrict__ didx, int E, long EpH,
                          float* __restrict__ a, float* __restrict__ den) {
  long i = (long)blockIdx.x * 256 + threadIdx.x;
  if (i >= EpH) return;
  int h = (int)(i % NHEAD);
  long e = i / NHEAD;
  int d = (e < E) ? didx[e] : (int)(e - E);
  float v = expf(logits[i] - fdec(mxu[d * NHEAD + h]));
  a[i] = v;
  atomicAdd(&den[d * NHEAD + h], v);
}

__global__ void gat_scatter_k(const float* __restrict__ a, const float* __restrict__ den,
                              const float* __restrict__ xl,
                              const int* __restrict__ sidx, const int* __restrict__ didx,
                              int E, int C, float* __restrict__ out) {
  long eh = blockIdx.x;
  int h = (int)(eh % NHEAD);
  long e = eh / NHEAD;
  int s, d;
  if (e < E) { s = sidx[e]; d = didx[e]; } else { s = d = (int)(e - E); }
  float alpha = a[eh] / den[d * NHEAD + h];
  int lane = threadIdx.x;
  const float* xs = xl + (size_t)s * NHEAD * C + (size_t)h * C;
  float* od = out + (size_t)d * NHEAD * C + (size_t)h * C;
  for (int cc = lane; cc < C; cc += 32) atomicAdd(&od[cc], alpha * xs[cc]);
}

// out[n, hc] += gat_bias[hc] + prev[n, hc % C]   (bias + tile(prev,(1,8)) residual)
__global__ void gat_finalize_k(float* __restrict__ out, const float* __restrict__ gbias,
                               const float* __restrict__ prev, int Nd, int C) {
  long i = (long)blockIdx.x * 256 + threadIdx.x;
  long tot = (long)Nd * NHEAD * C;
  if (i >= tot) return;
  int hc = (int)(i % (NHEAD * C));
  int n = (int)(i / (NHEAD * C));
  out[i] += gbias[hc] + prev[(size_t)n * C + hc % C];
}

// ---------------------------------------------------------------------------
// Output heads
// ---------------------------------------------------------------------------
__global__ void head_sum_k(const float* __restrict__ Hs, const float* __restrict__ Wv,
                           const float* __restrict__ bv, float* __restrict__ out) {
  int n = blockIdx.x;
  int lane = threadIdx.x;
  float s = 0.f;
  for (int c = lane; c < 128; c += 32) s += Hs[(size_t)n * 128 + c] * Wv[c];
  s = wave_reduce(s);
  if (lane == 0) out[n] = sigm(s + bv[0]);
}

__global__ void count_words_k(const int* __restrict__ w, int* __restrict__ c, int n) {
  int i = blockIdx.x * 256 + threadIdx.x;
  if (i < n) atomicAdd(&c[w[i]], 1);
}
__global__ void scan_serial_k(const int* __restrict__ c, int* __restrict__ o, int n) {
  if (blockIdx.x == 0 && threadIdx.x == 0) {
    int a = 0;
    for (int i = 0; i < n; ++i) { o[i] = a; a += c[i]; }
  }
}
__global__ void head_ner_k(const int* __restrict__ widx, const float* __restrict__ Hw,
                           const float* __restrict__ ea, const float* __restrict__ Wv,
                           const float* __restrict__ bv, const int* __restrict__ offs,
                           int* __restrict__ cur, float* __restrict__ out) {
  int e = blockIdx.x;
  int lane = threadIdx.x;
  int w = widx[e];
  float s = 0.f;
  for (int c = lane; c < 300; c += 32) s += Hw[(size_t)w * 300 + c] * Wv[c];
  for (int c = lane; c < 50; c += 32) s += ea[(size_t)e * 50 + c] * Wv[300 + c];
  s = wave_reduce(s);
  if (lane == 0) {
    int pos = offs[w] + atomicAdd(&cur[w], 1);  // argsort(word_idx) permutation
    out[pos] = sigm(s + bv[0]);
  }
}

// ---------------------------------------------------------------------------
// Host orchestration
// ---------------------------------------------------------------------------
#define GS(n) dim3((unsigned)((((long)(n)) + 255) / 256)), dim3(256), 0, stream

extern "C" void kernel_launch(void* const* d_in, const int* in_sizes, int n_in,
                              void* d_out, int out_size, void* d_ws, size_t ws_size,
                              hipStream_t stream) {
  (void)in_sizes; (void)n_in; (void)out_size; (void)ws_size;

  // ---- input pointers (setup_inputs dict order; params flattened in order) ----
  const int*   Xw       = (const int*)  d_in[0];
  const int*   Xs       = (const int*)  d_in[1];
  const int*   word_idx = (const int*)  d_in[2];
  const int*   sent_idx = (const int*)  d_in[3];
  const float* eraw     = (const float*)d_in[4];
  const float* word_emb = (const float*)d_in[5];
  const float* edge_emb = (const float*)d_in[6];
  const float* c1_w = (const float*)d_in[7];  const float* c1_b = (const float*)d_in[8];
  const float* bn1g = (const float*)d_in[9];  const float* bn1b = (const float*)d_in[10];
  const float* c2_w = (const float*)d_in[11]; const float* c2_b = (const float*)d_in[12];
  const float* bn2g = (const float*)d_in[13]; const float* bn2b = (const float*)d_in[14];
  const float* c3_w = (const float*)d_in[15]; const float* c3_b = (const float*)d_in[16];
  const float* bn3g = (const float*)d_in[17]; const float* bn3b = (const float*)d_in[18];
  const float* lf_Wih = (const float*)d_in[19]; const float* lf_Whh = (const float*)d_in[20];
  const float* lf_b   = (const float*)d_in[21];
  const float* lr_Wih = (const float*)d_in[22]; const float* lr_Whh = (const float*)d_in[23];
  const float* lr_b   = (const float*)d_in[24];
  const float* s2w_Wl = (const float*)d_in[25]; const float* s2w_bl = (const float*)d_in[26];
  const float* s2w_Wr = (const float*)d_in[27]; const float* s2w_br = (const float*)d_in[28];
  const float* s2w_We = (const float*)d_in[29]; const float* s2w_att = (const float*)d_in[30];
  const float* s2w_bias = (const float*)d_in[31];
  const float* w2s_Wl = (const float*)d_in[32]; const float* w2s_bl = (const float*)d_in[33];
  const float* w2s_Wr = (const float*)d_in[34]; const float* w2s_br = (const float*)d_in[35];
  const float* w2s_We = (const float*)d_in[36]; const float* w2s_att = (const float*)d_in[37];
  const float* w2s_bias = (const float*)d_in[38];
  const float* f1_W1 = (const float*)d_in[39]; const float* f1_b1 = (const float*)d_in[40];
  const float* f1_W2 = (const float*)d_in[41]; const float* f1_b2 = (const float*)d_in[42];
  const float* f2_W1 = (const float*)d_in[43]; const float* f2_b1 = (const float*)d_in[44];
  const float* f2_W2 = (const float*)d_in[45]; const float* f2_b2 = (const float*)d_in[46];
  const float* w_sum_W = (const float*)d_in[47]; const float* w_sum_b = (const float*)d_in[48];
  const float* w_ner_W = (const float*)d_in[49]; const float* w_ner_b = (const float*)d_in[50];
  float* dout = (float*)d_out;

  // ---- workspace bump allocator ----
  char* base = (char*)d_ws;
  size_t off = 0;
  auto alloc = [&](size_t bytes) -> void* {
    void* p = base + off;
    off = (off + bytes + 255) & ~(size_t)255;
    return p;
  };

  // Padded leading dims for bf16 GEMM staging (multiple of 32, zero-filled pad)
  const int LD300 = 320;

  // persistent buffers
  float* Hw     = (float*)alloc((size_t)NWRD * 300 * 4);
  float* Hs     = (float*)alloc((size_t)NSNT * 128 * 4);
  bf16*  Hw_bf  = (bf16*) alloc((size_t)NWRD * LD300 * 2);
  bf16*  Hs_bf  = (bf16*) alloc((size_t)NSNT * 128 * 2);
  float* eattr  = (float*)alloc((size_t)NEDG * 50 * 4);
  bf16* wb_lfWih = (bf16*)alloc((size_t)128 * LD300 * 2);
  bf16* wb_lrWih = (bf16*)alloc((size_t)128 * LD300 * 2);
  bf16* wb_s2wWl = (bf16*)alloc((size_t)2400 * 128 * 2);
  bf16* wb_s2wWr = (bf16*)alloc((size_t)2400 * LD300 * 2);
  bf16* wb_w2sWl = (bf16*)alloc((size_t)1024 * LD300 * 2);
  bf16* wb_w2sWr = (bf16*)alloc((size_t)1024 * 128 * 2);
  bf16* wb_f1W1  = (bf16*)alloc((size_t)512 * 2400 * 2);
  bf16* wb_f1W2  = (bf16*)alloc((size_t)300 * 512 * 2);
  bf16* wb_f2W1  = (bf16*)alloc((size_t)512 * 1024 * 2);
  bf16* wb_f2W2  = (bf16*)alloc((size_t)128 * 512 * 2);
  float*    bnm    = (float*)alloc(128 * 4);
  float*    bnv    = (float*)alloc(128 * 4);
  float*    logits = (float*)alloc((size_t)(NEDG + NSNT) * NHEAD * 4);
  float*    aexp   = (float*)alloc((size_t)(NEDG + NSNT) * NHEAD * 4);
  unsigned* mxu    = (unsigned*)alloc((size_t)NWRD * NHEAD * 4);
  float*    den    = (float*)alloc((size_t)NWRD * NHEAD * 4);
  int* counts = (int*)alloc(NWRD * 4);
  int* offs   = (int*)alloc(NWRD * 4);
  int* cursor = (int*)alloc(NWRD * 4);

  // arena (phase A: CNN/LSTM; phase B: GAT) — phases are disjoint in time
  size_t arena = off;
  auto carve = [&](size_t& o, size_t bytes) -> void* {
    void* p = base + o;
    o = (o + bytes + 255) & ~(size_t)255;
    return p;
  };
  size_t oA = arena;
  bf16*  Xse_bf = (bf16*) carve(oA, (size_t)NSNT * SLEN * LD300 * 2);
  float* Zin    = (float*)carve(oA, (size_t)NSNT * SLEN * 128 * 4);
  float* y1     = (float*)carve(oA, (size_t)NSNT * 100 * SLEN * 4);
  float* y2     = (float*)carve(oA, (size_t)NSNT * 100 * SLEN * 4);
  float* y3     = (float*)carve(oA, (size_t)NSNT * 64  * SLEN * 4);
  size_t oB = arena;
  float* xl_s2w = (float*)carve(oB, (size_t)NSNT * 2400 * 4);
  float* xr_s2w = (float*)carve(oB, (size_t)NWRD * 2400 * 4);  // reused as s2w GAT out
  bf16*  obf    = (bf16*) carve(oB, (size_t)NWRD * 2400 * 2);  // GEMM-A staging (bf16)
  float* xl_w2s = (float*)carve(oB, (size_t)NWRD * 1024 * 4);
  float* xr_w2s = (float*)carve(oB, (size_t)NSNT * 1024 * 4);  // reused as w2s GAT out
  float* hid    = (float*)carve(oB, (size_t)NWRD * 512 * 4);
  bf16*  hidbf  = (bf16*) carve(oB, (size_t)NWRD * 512 * 2);

  // ---- small host helpers ----
  auto cast_pad = [&](const float* s, bf16* d, int rows, int K, int ld) {
    cast_pad_k<<<GS((long)rows * ld)>>>(s, d, rows, K, ld);
  };
  auto gemm = [&](const bf16* A, int lda, const bf16* Wb, int ldw, const float* bias,
                  float* C, int M, int N, int K, bool relu) {
    int tm = (M + 15) >> 4;
    long tiles = (long)tm * ((N + 63) >> 6);
    unsigned blocks = (unsigned)((tiles + 7) >> 3);
    if (relu) gemm_bt_wmma<1><<<dim3(blocks), dim3(256), 0, stream>>>(A, lda, Wb, ldw, bias, C, M, N, K);
    else      gemm_bt_wmma<0><<<dim3(blocks), dim3(256), 0, stream>>>(A, lda, Wb, ldw, bias, C, M, N, K);
  };

  // ========== 1. weight bf16 staging (zero-padded K where needed) ==========
  cast_pad(lf_Wih, wb_lfWih, 128, 300, LD300);
  cast_pad(lr_Wih, wb_lrWih, 128, 300, LD300);
  cast_pad(s2w_Wl, wb_s2wWl, 2400, 128, 128);
  cast_pad(s2w_Wr, wb_s2wWr, 2400, 300, LD300);
  cast_pad(w2s_Wl, wb_w2sWl, 1024, 300, LD300);
  cast_pad(w2s_Wr, wb_w2sWr, 1024, 128, 128);
  cast_pad(f1_W1,  wb_f1W1,  512, 2400, 2400);
  cast_pad(f1_W2,  wb_f1W2,  300, 512, 512);
  cast_pad(f2_W1,  wb_f2W1,  512, 1024, 1024);
  cast_pad(f2_W2,  wb_f2W2,  128, 512, 512);

  // ========== 2. gathers ==========
  gather_rows_f32_k<<<GS((long)NWRD * 300)>>>(Xw, word_emb, Hw, NWRD, 300);
  gather_pad_bf16_k<<<GS((long)NSNT * SLEN * LD300)>>>(Xs, word_emb, Xse_bf,
                                                       NSNT * SLEN, 300, LD300);
  gather_edge_k<<<GS((long)NEDG * 50)>>>(eraw, edge_emb, eattr);

  // ========== 3. CNN -> Hs[:, 0:64] ==========
  conv1_embed_k<<<dim3(NSNT * SLEN), dim3(128), 0, stream>>>(Xs, word_emb, c1_w, c1_b, y1);
  bn_stats_k<<<dim3(100), dim3(256), 0, stream>>>(y1, NSNT, 100, SLEN, bnm, bnv);
  bn_relu_k<<<GS((long)NSNT * 100 * SLEN)>>>(y1, bnm, bnv, bn1g, bn1b, 100, SLEN, (long)NSNT * 100 * SLEN);
  conv1d_g_k<<<dim3(NSNT * SLEN), dim3(128), 0, stream>>>(y1, c2_w, c2_b, y2, NSNT, 100, 100, SLEN, 4, 1);
  bn_stats_k<<<dim3(100), dim3(256), 0, stream>>>(y2, NSNT, 100, SLEN, bnm, bnv);
  bn_relu_k<<<GS((long)NSNT * 100 * SLEN)>>>(y2, bnm, bnv, bn2g, bn2b, 100, SLEN, (long)NSNT * 100 * SLEN);
  conv1d_g_k<<<dim3(NSNT * SLEN), dim3(128), 0, stream>>>(y2, c3_w, c3_b, y3, NSNT, 100, 64, SLEN, 5, 2);
  bn_stats_k<<<dim3(64), dim3(256), 0, stream>>>(y3, NSNT, 64, SLEN, bnm, bnv);
  bn_relu_k<<<GS((long)NSNT * 64 * SLEN)>>>(y3, bnm, bnv, bn3g, bn3b, 64, SLEN, (long)NSNT * 64 * SLEN);
  xscnn_max_k<<<GS((long)NSNT * 64)>>>(y3, Hs);

  // ========== 4. BiLSTM -> Hs[:, 64:128] ==========
  gemm(Xse_bf, LD300, wb_lfWih, LD300, lf_b, Zin, NSNT * SLEN, 128, 300, false);
  lstm_k<<<dim3(NSNT), dim3(128), 0, stream>>>(Zin, lf_Whh, Xs, Hs, 0);
  gemm(Xse_bf, LD300, wb_lrWih, LD300, lr_b, Zin, NSNT * SLEN, 128, 300, false);
  lstm_k<<<dim3(NSNT), dim3(128), 0, stream>>>(Zin, lr_Whh, Xs, Hs, 1);

  // ========== 5. LayerNorm + bf16 activations ==========
  layernorm_k<<<dim3(NWRD), dim3(256), 0, stream>>>(Hw, 300);
  layernorm_k<<<dim3(NSNT), dim3(256), 0, stream>>>(Hs, 128);
  cast_pad(Hw, Hw_bf, NWRD, 300, LD300);
  cast_pad(Hs, Hs_bf, NSNT, 128, 128);

  // ---- GATv2 (host sequencer) ----
  auto run_gat = [&](int Ns, int Nd, int C,
                     const bf16* xsrc, int ldsrc, int Ksrc,
                     const bf16* xdst, int lddst, int Kdst,
                     const bf16* Wl, int ldWl, const float* bl,
                     const bf16* Wr, int ldWr, const float* br,
                     const float* We, const float* att, const float* gbias,
                     const float* prev, const int* sidx, const int* didx,
                     float* xl, float* xrout) {
    int HC = NHEAD * C;
    gemm(xsrc, ldsrc, Wl, ldWl, bl, xl, Ns, HC, Ksrc, false);
    gemm(xdst, lddst, Wr, ldWr, br, xrout, Nd, HC, Kdst, false);
    int nl = (Ns < Nd) ? Ns : Nd;
    long Ep = NEDG + nl;
    fill_key_k<<<GS((long)Nd * NHEAD)>>>(mxu, (long)Nd * NHEAD);
    fill_f32_k<<<GS((long)Nd * NHEAD)>>>(den, 0.f, (long)Nd * NHEAD);
    gat_logits_k<<<dim3((unsigned)(Ep * NHEAD)), dim3(32), 0, stream>>>(
        xl, xrout, eattr, We, att, sidx, didx, NEDG, C, logits, mxu);
    gat_exp_k<<<GS(Ep * NHEAD)>>>(logits, mxu, didx, NEDG, Ep * NHEAD, aexp, den);
    fill_f32_k<<<GS((long)Nd * HC)>>>(xrout, 0.f, (long)Nd * HC);  // xr dead: reuse as out
    gat_scatter_k<<<dim3((unsigned)(Ep * NHEAD)), dim3(32), 0, stream>>>(
        aexp, den, xl, sidx, didx, NEDG, C, xrout);
    gat_finalize_k<<<GS((long)Nd * HC)>>>(xrout, gbias, prev, Nd, C);
  };
  auto run_ffn_ln = [&](const float* x, int M, int Din, const bf16* W1b, const float* b1,
                        const bf16* W2b, const float* b2, int Dout,
                        float* out, bf16* out_bf, int out_ld) {
    cast_pad(x, obf, M, Din, Din);
    gemm(obf, Din, W1b, Din, b1, hid, M, 512, Din, true);
    cast_pad(hid, hidbf, M, 512, 512);
    gemm(hidbf, 512, W2b, 512, b2, out, M, Dout, 512, false);
    layernorm_k<<<dim3((unsigned)M), dim3(256), 0, stream>>>(out, Dout);
    cast_pad(out, out_bf, M, Dout, out_ld);
  };

  // ========== 6. 3 rounds of message passing ==========
  for (int it = 0; it < 3; ++it) {
    if (it != 0) {
      // sentence -> word: x_src=Hs(128), x_dst=Hw(300), C=300
      run_gat(NSNT, NWRD, 300,
              Hs_bf, 128, 128, Hw_bf, LD300, 300,
              wb_s2wWl, 128, s2w_bl, wb_s2wWr, LD300, s2w_br,
              s2w_We, s2w_att, s2w_bias,
              Hw, sent_idx, word_idx, xl_s2w, xr_s2w);
      run_ffn_ln(xr_s2w, NWRD, 2400, wb_f1W1, f1_b1, wb_f1W2, f1_b2, 300, Hw, Hw_bf, LD300);
    }
    // word -> sentence: x_src=Hw(300), x_dst=Hs(128), C=128
    run_gat(NWRD, NSNT, 128,
            Hw_bf, LD300, 300, Hs_bf, 128, 128,
            wb_w2sWl, LD300, w2s_bl, wb_w2sWr, 128, w2s_br,
            w2s_We, w2s_att, w2s_bias,
            Hs, word_idx, sent_idx, xl_w2s, xr_w2s);
    run_ffn_ln(xr_w2s, NSNT, 1024, wb_f2W1, f2_b1, wb_f2W2, f2_b2, 128, Hs, Hs_bf, 128);
  }

  // ========== 7. heads ==========
  head_sum_k<<<dim3(NSNT), dim3(32), 0, stream>>>(Hs, w_sum_W, w_sum_b, dout);

  fill_u32_k<<<GS((long)NWRD)>>>((unsigned*)counts, 0u, NWRD);
  fill_u32_k<<<GS((long)NWRD)>>>((unsigned*)cursor, 0u, NWRD);
  count_words_k<<<GS((long)NEDG)>>>(word_idx, counts, NEDG);
  scan_serial_k<<<dim3(1), dim3(1), 0, stream>>>(counts, offs, NWRD);
  head_ner_k<<<dim3(NEDG), dim3(32), 0, stream>>>(word_idx, Hw, eattr, w_ner_W, w_ner_b,
                                                  offs, cursor, dout + NSNT);
}
